// MultiHeadAttentionBlock_63290638074365
// MI455X (gfx1250) — matrix operand order repaired and checked by reading, not compiled, not run on hardware
//
#include <hip/hip_runtime.h>
#include <hip/hip_bf16.h>

typedef __bf16 bf16_t;
typedef __attribute__((ext_vector_type(16))) __bf16   v16bf;
typedef __attribute__((ext_vector_type(8)))  float    v8f;
typedef __attribute__((ext_vector_type(4)))  unsigned u32x4;
typedef __attribute__((ext_vector_type(4)))  float    f32x4;

#define D_MODEL 2048
#define N_HEADS 16
#define DK      128
#define BATCH   2
#define SEQ     2048
#define MROWS   (BATCH * SEQ)   // 4096

__device__ __forceinline__ bf16_t f2b(float x) {
    union { float f; unsigned u; } a; a.f = x;
    unsigned r = (a.u + 0x7fffu + ((a.u >> 16) & 1u)) >> 16;
    union { unsigned short s; bf16_t b; } c; c.s = (unsigned short)r;
    return c.b;
}

__device__ __forceinline__ v8f zero_v8() {
    v8f v;
    #pragma unroll
    for (int i = 0; i < 8; ++i) v[i] = 0.0f;
    return v;
}

union FR { u32x4 q[2]; v16bf v; };

// ---------------------------------------------------------------------------
// GEMM: Y[M,N] = X[M,K] @ W[N,K]^T + bias[N].  fp32 in, bf16 compute via WMMA,
// fp32 accumulate.  Optional fp32 and/or bf16 outputs.
// Block tile 128x128, 8 waves, each wave 2x4 tiles of 16x16, K-step 32.
// ---------------------------------------------------------------------------
__global__ __launch_bounds__(256)
void gemm_xwt(const float* __restrict__ A, const float* __restrict__ W,
              const float* __restrict__ bias, float* __restrict__ Yf,
              bf16_t* __restrict__ Yb, int M, int N, int K)
{
    __shared__ __align__(16) bf16_t sA[128 * 40];  // row stride 80 B
    __shared__ __align__(16) bf16_t sW[128 * 40];

    const int tid  = threadIdx.x;
    const int lane = tid & 31;
    const int wave = tid >> 5;
    const bool lh  = (lane >= 16);
    const int nlo  = lane & 15;
    const int m0 = blockIdx.y * 128;
    const int n0 = blockIdx.x * 128;
    const int wm = (wave >> 1) * 32;   // 0,32,64,96
    const int wn = (wave & 1) * 64;    // 0,64

    v8f acc[2][4];
    #pragma unroll
    for (int mt = 0; mt < 2; ++mt)
        #pragma unroll
        for (int nt = 0; nt < 4; ++nt) acc[mt][nt] = zero_v8();

    const int ldRow  = tid >> 1;   // 0..127
    const int ldHalf = tid & 1;    // 0..1  (16 elements each)
    const float* aSrc = A + (size_t)(m0 + ldRow) * K + ldHalf * 16;
    const float* wSrc = W + (size_t)(n0 + ldRow) * K + ldHalf * 16;
    char* sAdst = (char*)sA + ldRow * 80 + ldHalf * 32;
    char* sWdst = (char*)sW + ldRow * 80 + ldHalf * 32;

    for (int k0 = 0; k0 < K; k0 += 32) {
        if (k0 + 32 < K) {
            __builtin_prefetch(aSrc + k0 + 32, 0, 1);
            __builtin_prefetch(wSrc + k0 + 32, 0, 1);
        }
        __syncthreads();
        {   // stage 128x32 fp32 tiles -> bf16 LDS
            const f32x4* a4 = (const f32x4*)(aSrc + k0);
            const f32x4* w4 = (const f32x4*)(wSrc + k0);
            f32x4 af[4], wf[4];
            #pragma unroll
            for (int i = 0; i < 4; ++i) { af[i] = a4[i]; wf[i] = w4[i]; }
            union { bf16_t h[16]; u32x4 q[2]; } pa, pw;
            #pragma unroll
            for (int i = 0; i < 16; ++i) {
                pa.h[i] = f2b(af[i >> 2][i & 3]);
                pw.h[i] = f2b(wf[i >> 2][i & 3]);
            }
            ((u32x4*)sAdst)[0] = pa.q[0]; ((u32x4*)sAdst)[1] = pa.q[1];
            ((u32x4*)sWdst)[0] = pw.q[0]; ((u32x4*)sWdst)[1] = pw.q[1];
        }
        __syncthreads();

        FR afr[2], bfr[4];
        #pragma unroll
        for (int mt = 0; mt < 2; ++mt) {     // A frag: K split {0-7,16-23}/{8-15,24-31}
            int row = wm + mt * 16 + nlo;
            const char* p = (const char*)sA + row * 80 + (lh ? 16 : 0);
            afr[mt].q[0] = *(const u32x4*)p;
            afr[mt].q[1] = *(const u32x4*)(p + 32);
        }
        #pragma unroll
        for (int nt = 0; nt < 4; ++nt) {     // B frag: K split {0-7,16-23}/{8-15,24-31}
            int row = wn + nt * 16 + nlo;
            const char* p = (const char*)sW + row * 80 + (lh ? 16 : 0);
            bfr[nt].q[0] = *(const u32x4*)p;
            bfr[nt].q[1] = *(const u32x4*)(p + 32);
        }
        #pragma unroll
        for (int mt = 0; mt < 2; ++mt)
            #pragma unroll
            for (int nt = 0; nt < 4; ++nt)
                acc[mt][nt] = __builtin_amdgcn_wmma_f32_16x16x32_bf16(
                    false, afr[mt].v, false, bfr[nt].v, (short)0, acc[mt][nt], false, false);
    }

    #pragma unroll
    for (int mt = 0; mt < 2; ++mt)
        #pragma unroll
        for (int nt = 0; nt < 4; ++nt) {
            int col = n0 + wn + nt * 16 + nlo;
            float bv = bias[col];
            #pragma unroll
            for (int e = 0; e < 8; ++e) {
                int row = m0 + wm + mt * 16 + e + (lh ? 8 : 0);
                float v = acc[mt][nt][e] + bv;
                if (Yf) Yf[(size_t)row * N + col] = v;
                if (Yb) Yb[(size_t)row * N + col] = f2b(v);
            }
        }
}

// ---------------------------------------------------------------------------
// Flash attention with bug-faithful reshape:
//   qv[b,h,16I+ii,d] = Q[b, h*128+I, ii*128+d]  (same for K,V; inverse on out)
// One block per (b,h, 128-row i-block); wave w owns s-tile I = blk*8+w.
// Online softmax with floor(s/sqrt(2048)) and causal mask.
// K tile staged via GLOBAL_LOAD_ASYNC_TO_LDS_B128 (ASYNCcnt path).
// ---------------------------------------------------------------------------
__global__ __launch_bounds__(256)
void attn_kernel(const bf16_t* __restrict__ Qb, const bf16_t* __restrict__ Kb,
                 const bf16_t* __restrict__ Vb, float* __restrict__ Xo)
{
    __shared__ __align__(16) bf16_t sK[32 * 136];    // [2 j-tiles][16 jj][128 d], 272 B stride
    __shared__ __align__(16) bf16_t sV[128 * 40];    // V^T: [128 d][32 j], 80 B stride
    __shared__ __align__(16) bf16_t sP[8 * 16 * 40]; // per-wave P tile [16 i][32 j]

    const int tid  = threadIdx.x;
    const int lane = tid & 31;
    const int wave = tid >> 5;
    const bool lh  = (lane >= 16);
    const int nlo  = lane & 15;
    const int blk = blockIdx.x;     // 0..15  (128 i rows each)
    const int h   = blockIdx.y;
    const int b   = blockIdx.z;
    const int I   = blk * 8 + wave; // s-tile 0..127
    const size_t row = (size_t)b * SEQ + h * DK + I;   // original row index
    const bf16_t* qrow = Qb + row * D_MODEL;

    // Preload 4 Q A-fragments (whole 16x128 tile lives in one 2048-elem row).
    FR qf[4];
    #pragma unroll
    for (int ds = 0; ds < 4; ++ds) {
        int base = nlo * 128 + ds * 32 + (lh ? 8 : 0);
        qf[ds].q[0] = *(const u32x4*)(qrow + base);
        qf[ds].q[1] = *(const u32x4*)(qrow + base + 16);
    }

    float m_[8], l_[8];
    v8f out[8];
    #pragma unroll
    for (int e = 0; e < 8; ++e) { m_[e] = -3.0e38f; l_[e] = 0.0f; }
    #pragma unroll
    for (int t = 0; t < 8; ++t) out[t] = zero_v8();

    const int iBase = I * 16;
    const int jcMax = (blk + 1) * 4;   // causal: chunks of 32 j's
    const float rs = 0.022097086912079608f;  // 1/sqrt(2048)

    const int jt_ld = tid >> 7;          // which of 2 j-tiles this thread stages
    const int o_ld  = (tid & 127) * 16;  // offset within the 2048-elem source row
    const int jj_ld = o_ld >> 7;
    const int dl_ld = o_ld & 127;

    // LDS raw byte offset for this thread's K destination (flat addr low bits).
    char* kd = (char*)sK + (jt_ld * 16 + jj_ld) * 272 + dl_ld * 2;
    const unsigned ldsK = (unsigned)(uintptr_t)kd;

    for (int jc = 0; jc < jcMax; ++jc) {
        const int J0 = jc * 2;
        __syncthreads();
        {   // cooperative staging: K via async global->LDS DMA, V transposed via VALU
            const bf16_t* krow = Kb + ((size_t)b * SEQ + h * DK + J0 + jt_ld) * D_MODEL;
            const bf16_t* vrow = Vb + ((size_t)b * SEQ + h * DK + J0 + jt_ld) * D_MODEL;
            const bf16_t* kg = krow + o_ld;
            asm volatile("global_load_async_to_lds_b128 %0, %1, off"
                         :: "v"(ldsK), "v"(kg) : "memory");
            asm volatile("global_load_async_to_lds_b128 %0, %1, off offset:16"
                         :: "v"(ldsK), "v"(kg) : "memory");
            union { u32x4 q[2]; bf16_t hh[16]; } vv;
            vv.q[0] = *(const u32x4*)(vrow + o_ld);
            vv.q[1] = *(const u32x4*)(vrow + o_ld + 8);
            const int jcl = jt_ld * 16 + jj_ld;
            #pragma unroll
            for (int e = 0; e < 16; ++e)
                sV[(dl_ld + e) * 40 + jcl] = vv.hh[e];
            asm volatile("s_wait_asynccnt 0" ::: "memory");
        }
        __syncthreads();

        if (jc * 32 <= iBase + 15) {     // wave-uniform causal skip
            v8f s0 = zero_v8(), s1 = zero_v8();
            #pragma unroll
            for (int jt = 0; jt < 2; ++jt) {
                #pragma unroll
                for (int ds = 0; ds < 4; ++ds) {
                    FR kf;
                    const char* p = (const char*)sK + (jt * 16 + nlo) * 272
                                    + ds * 64 + (lh ? 16 : 0);
                    kf.q[0] = *(const u32x4*)p;
                    kf.q[1] = *(const u32x4*)(p + 32);
                    if (jt == 0)
                        s0 = __builtin_amdgcn_wmma_f32_16x16x32_bf16(
                            false, qf[ds].v, false, kf.v, (short)0, s0, false, false);
                    else
                        s1 = __builtin_amdgcn_wmma_f32_16x16x32_bf16(
                            false, qf[ds].v, false, kf.v, (short)0, s1, false, false);
                }
            }

            const int jcol0 = jc * 32 + nlo;
            const int jcol1 = jcol0 + 16;
            float p0[8], p1[8];
            #pragma unroll
            for (int e = 0; e < 8; ++e) {
                const int ig = iBase + e + (lh ? 8 : 0);
                float a = floorf(s0[e] * rs);
                float c = floorf(s1[e] * rs);
                if (jcol0 > ig) a = -1.0e9f;
                if (jcol1 > ig) c = -1.0e9f;
                float rm = fmaxf(a, c);
                rm = fmaxf(rm, __shfl_xor(rm, 1, 32));
                rm = fmaxf(rm, __shfl_xor(rm, 2, 32));
                rm = fmaxf(rm, __shfl_xor(rm, 4, 32));
                rm = fmaxf(rm, __shfl_xor(rm, 8, 32));
                float mn = fmaxf(m_[e], rm);
                float corr = __expf(m_[e] - mn);
                float e0 = __expf(a - mn);
                float e1 = __expf(c - mn);
                float rsum = e0 + e1;
                rsum += __shfl_xor(rsum, 1, 32);
                rsum += __shfl_xor(rsum, 2, 32);
                rsum += __shfl_xor(rsum, 4, 32);
                rsum += __shfl_xor(rsum, 8, 32);
                l_[e] = l_[e] * corr + rsum;
                m_[e] = mn;
                #pragma unroll
                for (int t = 0; t < 8; ++t) out[t][e] *= corr;
                p0[e] = e0; p1[e] = e1;
            }

            // C-layout P -> A-layout via wave-private LDS
            char* pb = (char*)sP + wave * 16 * 80;
            #pragma unroll
            for (int e = 0; e < 8; ++e) {
                const int r = e + (lh ? 8 : 0);
                *(bf16_t*)(pb + r * 80 + nlo * 2)        = f2b(p0[e]);
                *(bf16_t*)(pb + r * 80 + (16 + nlo) * 2) = f2b(p1[e]);
            }
            FR pf;
            {
                const char* p = pb + nlo * 80 + (lh ? 16 : 0);
                pf.q[0] = *(const u32x4*)p;
                pf.q[1] = *(const u32x4*)(p + 32);
            }
            #pragma unroll
            for (int dt = 0; dt < 8; ++dt) {
                FR vf;
                const char* p = (const char*)sV + (dt * 16 + nlo) * 80 + (lh ? 16 : 0);
                vf.q[0] = *(const u32x4*)p;
                vf.q[1] = *(const u32x4*)(p + 32);
                out[dt] = __builtin_amdgcn_wmma_f32_16x16x32_bf16(
                    false, pf.v, false, vf.v, (short)0, out[dt], false, false);
            }
        }
    }

    // normalize + inverse-chop store (fp32 for the final projection GEMM)
    float* orow = Xo + row * D_MODEL;
    #pragma unroll
    for (int e = 0; e < 8; ++e) {
        const float inv = (l_[e] > 0.0f) ? (1.0f / l_[e]) : 0.0f;
        const int ii = e + (lh ? 8 : 0);
        #pragma unroll
        for (int dt = 0; dt < 8; ++dt)
            orow[ii * 128 + dt * 16 + nlo] = out[dt][e] * inv;
    }
}

extern "C" void kernel_launch(void* const* d_in, const int* in_sizes, int n_in,
                              void* d_out, int out_size, void* d_ws, size_t ws_size,
                              hipStream_t stream)
{
    (void)in_sizes; (void)n_in; (void)out_size; (void)ws_size;
    const float* key   = (const float*)d_in[0];
    const float* query = (const float*)d_in[1];
    const float* value = (const float*)d_in[2];
    // d_in[3] = mask (causal tril, hardcoded in the attention kernel)
    const float* Wq = (const float*)d_in[4];
    const float* bq = (const float*)d_in[5];
    const float* Wk = (const float*)d_in[6];
    const float* bk = (const float*)d_in[7];
    const float* Wv = (const float*)d_in[8];
    const float* bv = (const float*)d_in[9];
    const float* Wo = (const float*)d_in[10];
    const float* bo = (const float*)d_in[11];

    const size_t nElems = (size_t)MROWS * D_MODEL;  // 8M
    bf16_t* Qb = (bf16_t*)d_ws;
    bf16_t* Kb = Qb + nElems;
    bf16_t* Vb = Kb + nElems;
    float*  Xo = (float*)(Vb + nElems);

    dim3 gg(D_MODEL / 128, MROWS / 128);
    gemm_xwt<<<gg, 256, 0, stream>>>(query, Wq, bq, nullptr, Qb, MROWS, D_MODEL, D_MODEL);
    gemm_xwt<<<gg, 256, 0, stream>>>(key,   Wk, bk, nullptr, Kb, MROWS, D_MODEL, D_MODEL);
    gemm_xwt<<<gg, 256, 0, stream>>>(value, Wv, bv, nullptr, Vb, MROWS, D_MODEL, D_MODEL);

    dim3 ga(SEQ / 128, N_HEADS, BATCH);
    attn_kernel<<<ga, 256, 0, stream>>>(Qb, Kb, Vb, Xo);

    gemm_xwt<<<gg, 256, 0, stream>>>(Xo, Wo, bo, (float*)d_out, nullptr, MROWS, D_MODEL, D_MODEL);
}
